// LSTMGRUHybrid_31207232373043
// MI455X (gfx1250) — matrix operand order repaired and checked by reading, not compiled, not run on hardware
//
#include <hip/hip_runtime.h>

// ---------------------------------------------------------------------------
// LSTM+GRU hybrid, persistent-RNN style for MI455X (gfx1250, wave32, WMMA).
// All GEMMs use v_wmma_f32_16x16x32_bf16 (bf16 A/B, fp32 accumulate).
// One workgroup scans T=512 steps for a 16-row batch tile; recurrent weights
// live in VGPRs as pre-built B-fragments, so each step is only LDS traffic
// + 16 WMMAs per wave.
// ---------------------------------------------------------------------------

typedef __attribute__((ext_vector_type(16))) __bf16 v16bf;
typedef __attribute__((ext_vector_type(8)))  __bf16 v8bf;
typedef __attribute__((ext_vector_type(8)))  float  v8f;

#define B_     512
#define T_     512
#define D_     128
#define H_     128
#define FOURH  512
#define G_     128
#define THREEG 384
#define F_     160
#define C_     64
#define LN_EPS 1e-5f

__device__ __forceinline__ float sigmoidf_(float x) { return 1.0f / (1.0f + __expf(-x)); }

__device__ __forceinline__ v8f wmma_bf16(v16bf a, v16bf b, v8f c) {
  // D = A(16x32 bf16) * B(32x16 bf16) + C(16x16 f32)
  return __builtin_amdgcn_wmma_f32_16x16x32_bf16(false, a, false, b, (short)0, c, false, false);
}

// A-fragment (16x32 bf16) from a row-major LDS tile [16][128] bf16.
// ISA layout: lanes 0-15 hold row M=lane, K in [kc*32+0,8)U[kc*32+16,24);
// lanes 16-31 hold the same rows with K shifted by +8. -> two b128 LDS loads.
__device__ __forceinline__ v16bf load_A_lds(const __bf16* tile, int lane, int kc) {
  const int m = lane & 15, grp = lane >> 4;
  const __bf16* p = tile + m * 128 + kc * 32 + grp * 8;
  v8bf lo = *(const v8bf*)p;
  v8bf hi = *(const v8bf*)(p + 16);
  v16bf r;
#pragma unroll
  for (int i = 0; i < 8; ++i) { r[i] = lo[i]; r[i + 8] = hi[i]; }
  return r;
}

// B-fragment (32x16 bf16, i.e. W^T tile) straight from row-major fp32 weights
// W[out_features][in_features]: lane (n = lane&15, grp = lane>>4) needs K in
// [kc*32+grp*8, +8) U [kc*32+16+grp*8, +8) of row (ncol0+n) (matches A).
__device__ __forceinline__ v16bf load_B_gl(const float* W, int ldk, int ncol0, int lane, int kc) {
  const int n = lane & 15, grp = lane >> 4;
  const float* p = W + (size_t)(ncol0 + n) * ldk + kc * 32 + grp * 8;
  v16bf r;
#pragma unroll
  for (int j = 0; j < 16; ++j) r[j] = (__bf16)p[j < 8 ? j : j + 8];
  return r;
}

// ---------------------------------------------------------------------------
// Kernel 1: fused LSTM scan + LayerNorm. 32 WGs x (16 waves = 512 thr).
// Each wave owns a 32-column slice of the 4H=512 gate dim; Wih^T/Whh^T
// fragments (2 ntiles x 4 kchunks each) stay in VGPRs for all 512 steps.
// Emits post-LN lstm_out as bf16 [B][T][H] (the GRU's A operand).
// ---------------------------------------------------------------------------
__global__ __launch_bounds__(512, 1)
void lstm_scan_kernel(const float* __restrict__ x,
                      const float* __restrict__ Wih, const float* __restrict__ Whh,
                      const float* __restrict__ bih, const float* __restrict__ bhh,
                      const float* __restrict__ lng, const float* __restrict__ lnb,
                      __bf16* __restrict__ out_bf) {
  __shared__ __bf16 x_lds[16][128];   // x_t tile, bf16
  __shared__ __bf16 h_lds[16][128];   // h_{t-1} tile, bf16
  __shared__ float  g_lds[16][FOURH]; // gate pre-activations, fp32

  const int tid = threadIdx.x;
  const int wave = tid >> 5;
  const int lane = tid & 31;
  const int grp  = lane >> 4;
  const int b0   = blockIdx.x * 16;
  const int gc0  = lane * 4;          // this thread's 4 hidden columns

  // Weight fragments in registers (loaded once).
  v16bf wih[2][4], whh[2][4];
  float bias[2];
#pragma unroll
  for (int nt = 0; nt < 2; ++nt) {
    const int nc = wave * 32 + nt * 16;
#pragma unroll
    for (int kc = 0; kc < 4; ++kc) {
      wih[nt][kc] = load_B_gl(Wih, D_, nc, lane, kc);
      whh[nt][kc] = load_B_gl(Whh, H_, nc, lane, kc);
    }
    const int col = nc + (lane & 15);
    bias[nt] = bih[col] + bhh[col];
  }

  float lngv[4], lnbv[4], c_reg[4];
#pragma unroll
  for (int j = 0; j < 4; ++j) { lngv[j] = lng[gc0 + j]; lnbv[j] = lnb[gc0 + j]; c_reg[j] = 0.f; }

  *(uint2*)&h_lds[wave][gc0] = make_uint2(0u, 0u);  // h_0 = 0

  const float* xrow = x + (size_t)(b0 + wave) * T_ * D_ + gc0;

  for (int t = 0; t < T_; ++t) {
    // Stage x_t: wave r stages row r, 4 floats/lane -> bf16.
    const float4 xv = *(const float4*)(xrow + (size_t)t * D_);
    x_lds[wave][gc0 + 0] = (__bf16)xv.x;
    x_lds[wave][gc0 + 1] = (__bf16)xv.y;
    x_lds[wave][gc0 + 2] = (__bf16)xv.z;
    x_lds[wave][gc0 + 3] = (__bf16)xv.w;
    if (t + 1 < T_) __builtin_prefetch(xrow + (size_t)(t + 1) * D_, 0, 3);
    __syncthreads();

    // g = x_t @ Wih^T + h @ Whh^T + (bih + bhh), for this wave's 32 columns.
    v16bf ax[4], ah[4];
#pragma unroll
    for (int kc = 0; kc < 4; ++kc) {
      ax[kc] = load_A_lds(&x_lds[0][0], lane, kc);
      ah[kc] = load_A_lds(&h_lds[0][0], lane, kc);
    }
#pragma unroll
    for (int nt = 0; nt < 2; ++nt) {
      v8f acc = {};
#pragma unroll
      for (int kc = 0; kc < 4; ++kc) {
        acc = wmma_bf16(ax[kc], wih[nt][kc], acc);
        acc = wmma_bf16(ah[kc], whh[nt][kc], acc);
      }
      const int ncol = wave * 32 + nt * 16 + (lane & 15);
#pragma unroll
      for (int i = 0; i < 8; ++i) g_lds[grp * 8 + i][ncol] = acc[i] + bias[nt];
    }
    __syncthreads();

    // Gate math (PyTorch order i,f,g,o): thread owns row=wave, cols gc0..+3.
    float hv[4], s = 0.f, s2 = 0.f;
#pragma unroll
    for (int j = 0; j < 4; ++j) {
      const int cc = gc0 + j;
      const float gi = g_lds[wave][cc];
      const float gf = g_lds[wave][128 + cc];
      const float gg = g_lds[wave][256 + cc];
      const float go = g_lds[wave][384 + cc];
      c_reg[j] = sigmoidf_(gf) * c_reg[j] + sigmoidf_(gi) * tanhf(gg);
      const float h = sigmoidf_(go) * tanhf(c_reg[j]);
      hv[j] = h; s += h; s2 += h * h;
    }
    // LayerNorm over H=128: all 32 lanes of this wave hold the same row.
#pragma unroll
    for (int m = 1; m < 32; m <<= 1) { s += __shfl_xor(s, m, 32); s2 += __shfl_xor(s2, m, 32); }
    const float mu   = s * (1.f / 128.f);
    const float rstd = rsqrtf(s2 * (1.f / 128.f) - mu * mu + LN_EPS);

    uint2 pk; __bf16* pp = (__bf16*)&pk;
#pragma unroll
    for (int j = 0; j < 4; ++j) {
      h_lds[wave][gc0 + j] = (__bf16)hv[j];  // raw h feeds the recurrence
      pp[j] = (__bf16)((hv[j] - mu) * rstd * lngv[j] + lnbv[j]);
    }
    *(uint2*)(out_bf + ((size_t)(b0 + wave) * T_ + t) * H_ + gc0) = pk;
    __syncthreads();
  }
}

// ---------------------------------------------------------------------------
// Kernel 2: fused GRU scan over LN(lstm_out); emits relu(LN(h_T)) [B][G] f32.
// 3G=384 gate columns -> waves 0..11 do WMMA (2 ntiles each); all 16 waves
// stage/gate. gx and gh kept separate (n-gate needs r*ghn).
// ---------------------------------------------------------------------------
__global__ __launch_bounds__(512, 1)
void gru_scan_kernel(const __bf16* __restrict__ xin,  // [B][T][H] bf16
                     const float* __restrict__ Wih, const float* __restrict__ Whh,
                     const float* __restrict__ bih, const float* __restrict__ bhh,
                     const float* __restrict__ lng, const float* __restrict__ lnb,
                     float* __restrict__ h_last) {
  __shared__ __bf16 x_lds[16][128];
  __shared__ __bf16 h_lds[16][128];
  __shared__ float  gx_lds[16][THREEG];
  __shared__ float  gh_lds[16][THREEG];

  const int tid = threadIdx.x;
  const int wave = tid >> 5;
  const int lane = tid & 31;
  const int grp  = lane >> 4;
  const int b0   = blockIdx.x * 16;
  const int gc0  = lane * 4;
  const bool mma_wave = (wave < 12);  // 12 waves * 32 cols = 384

  v16bf gih[2][4], ghh[2][4];
  float bx[2] = {0.f, 0.f}, bh[2] = {0.f, 0.f};
  if (mma_wave) {
#pragma unroll
    for (int nt = 0; nt < 2; ++nt) {
      const int nc = wave * 32 + nt * 16;
#pragma unroll
      for (int kc = 0; kc < 4; ++kc) {
        gih[nt][kc] = load_B_gl(Wih, H_, nc, lane, kc);
        ghh[nt][kc] = load_B_gl(Whh, G_, nc, lane, kc);
      }
      const int col = nc + (lane & 15);
      bx[nt] = bih[col]; bh[nt] = bhh[col];
    }
  }

  float h_reg[4] = {0.f, 0.f, 0.f, 0.f};
  float lngv[4], lnbv[4];
#pragma unroll
  for (int j = 0; j < 4; ++j) { lngv[j] = lng[gc0 + j]; lnbv[j] = lnb[gc0 + j]; }
  *(uint2*)&h_lds[wave][gc0] = make_uint2(0u, 0u);

  const __bf16* xrow = xin + (size_t)(b0 + wave) * T_ * H_ + gc0;

  for (int t = 0; t < T_; ++t) {
    *(uint2*)&x_lds[wave][gc0] = *(const uint2*)(xrow + (size_t)t * H_);
    if (t + 1 < T_) __builtin_prefetch(xrow + (size_t)(t + 1) * H_, 0, 3);
    __syncthreads();

    if (mma_wave) {
      v16bf ax[4], ah[4];
#pragma unroll
      for (int kc = 0; kc < 4; ++kc) {
        ax[kc] = load_A_lds(&x_lds[0][0], lane, kc);
        ah[kc] = load_A_lds(&h_lds[0][0], lane, kc);
      }
#pragma unroll
      for (int nt = 0; nt < 2; ++nt) {
        v8f accx = {}, acch = {};
#pragma unroll
        for (int kc = 0; kc < 4; ++kc) {
          accx = wmma_bf16(ax[kc], gih[nt][kc], accx);
          acch = wmma_bf16(ah[kc], ghh[nt][kc], acch);
        }
        const int ncol = wave * 32 + nt * 16 + (lane & 15);
#pragma unroll
        for (int i = 0; i < 8; ++i) {
          gx_lds[grp * 8 + i][ncol] = accx[i] + bx[nt];
          gh_lds[grp * 8 + i][ncol] = acch[i] + bh[nt];
        }
      }
    }
    __syncthreads();

    // Gate math (PyTorch order r,z,n): row=wave, cols gc0..+3.
#pragma unroll
    for (int j = 0; j < 4; ++j) {
      const int cc = gc0 + j;
      const float r = sigmoidf_(gx_lds[wave][cc]       + gh_lds[wave][cc]);
      const float z = sigmoidf_(gx_lds[wave][128 + cc] + gh_lds[wave][128 + cc]);
      const float n = tanhf(gx_lds[wave][256 + cc] + r * gh_lds[wave][256 + cc]);
      h_reg[j] = (1.f - z) * n + z * h_reg[j];
      h_lds[wave][cc] = (__bf16)h_reg[j];
    }
    __syncthreads();
  }

  // Final LN over G=128 + ReLU.
  float s = 0.f, s2 = 0.f;
#pragma unroll
  for (int j = 0; j < 4; ++j) { s += h_reg[j]; s2 += h_reg[j] * h_reg[j]; }
#pragma unroll
  for (int m = 1; m < 32; m <<= 1) { s += __shfl_xor(s, m, 32); s2 += __shfl_xor(s2, m, 32); }
  const float mu   = s * (1.f / 128.f);
  const float rstd = rsqrtf(s2 * (1.f / 128.f) - mu * mu + LN_EPS);
#pragma unroll
  for (int j = 0; j < 4; ++j) {
    const float y = (h_reg[j] - mu) * rstd * lngv[j] + lnbv[j];
    h_last[(size_t)(b0 + wave) * G_ + gc0 + j] = fmaxf(y, 0.f);
  }
}

// ---------------------------------------------------------------------------
// Kernel 3: tiny MLP head (<0.05% of total FLOPs) — plain VALU.
// z = relu(LN(h @ fc1_W^T + fc1_b)); out = z @ fco_W^T + fco_b.
// ---------------------------------------------------------------------------
__global__ __launch_bounds__(256)
void head_kernel(const float* __restrict__ h_last,
                 const float* __restrict__ fc1_W, const float* __restrict__ fc1_b,
                 const float* __restrict__ lng, const float* __restrict__ lnb,
                 const float* __restrict__ fco_W, const float* __restrict__ fco_b,
                 float* __restrict__ out) {
  __shared__ float hrow[G_];
  __shared__ float z1[F_];
  __shared__ float stats[2];
  const int b = blockIdx.x, tid = threadIdx.x;

  if (tid < G_) hrow[tid] = h_last[(size_t)b * G_ + tid];
  __syncthreads();

  if (tid < F_) {
    const float* w = fc1_W + (size_t)tid * G_;
    float s = fc1_b[tid];
#pragma unroll 8
    for (int k = 0; k < G_; ++k) s += hrow[k] * w[k];
    z1[tid] = s;
  }
  __syncthreads();

  if (tid < 32) {
    float s = 0.f, s2 = 0.f;
    for (int j = tid; j < F_; j += 32) { const float v = z1[j]; s += v; s2 += v * v; }
#pragma unroll
    for (int m = 1; m < 32; m <<= 1) { s += __shfl_xor(s, m, 32); s2 += __shfl_xor(s2, m, 32); }
    if (tid == 0) {
      const float mu = s / (float)F_;
      stats[0] = mu;
      stats[1] = rsqrtf(s2 / (float)F_ - mu * mu + LN_EPS);
    }
  }
  __syncthreads();

  if (tid < F_) {
    const float y = (z1[tid] - stats[0]) * stats[1] * lng[tid] + lnb[tid];
    z1[tid] = fmaxf(y, 0.f);
  }
  __syncthreads();

  if (tid < C_) {
    const float* w = fco_W + (size_t)tid * F_;
    float s = fco_b[tid];
#pragma unroll 8
    for (int k = 0; k < F_; ++k) s += z1[k] * w[k];
    out[(size_t)b * C_ + tid] = s;
  }
}

// ---------------------------------------------------------------------------
extern "C" void kernel_launch(void* const* d_in, const int* in_sizes, int n_in,
                              void* d_out, int out_size, void* d_ws, size_t ws_size,
                              hipStream_t stream) {
  const float* x         = (const float*)d_in[0];
  const float* lstm_Wih  = (const float*)d_in[1];
  const float* lstm_Whh  = (const float*)d_in[2];
  const float* lstm_bih  = (const float*)d_in[3];
  const float* lstm_bhh  = (const float*)d_in[4];
  const float* gru_Wih   = (const float*)d_in[5];
  const float* gru_Whh   = (const float*)d_in[6];
  const float* gru_bih   = (const float*)d_in[7];
  const float* gru_bhh   = (const float*)d_in[8];
  const float* ln_lstm_g = (const float*)d_in[9];
  const float* ln_lstm_b = (const float*)d_in[10];
  const float* ln_gru_g  = (const float*)d_in[11];
  const float* ln_gru_b  = (const float*)d_in[12];
  const float* fc1_W     = (const float*)d_in[13];
  const float* fc1_b     = (const float*)d_in[14];
  const float* ln_fc1_g  = (const float*)d_in[15];
  const float* ln_fc1_b  = (const float*)d_in[16];
  const float* fco_W     = (const float*)d_in[17];
  const float* fco_b     = (const float*)d_in[18];
  float* out = (float*)d_out;

  // Workspace layout: [0, 64MB) lstm_out bf16 [B][T][H]; then h_last f32 [B][G].
  char* ws = (char*)d_ws;
  __bf16* lstm_bf = (__bf16*)ws;
  float*  h_last  = (float*)(ws + (size_t)B_ * T_ * H_ * sizeof(__bf16));

  lstm_scan_kernel<<<B_ / 16, 512, 0, stream>>>(
      x, lstm_Wih, lstm_Whh, lstm_bih, lstm_bhh, ln_lstm_g, ln_lstm_b, lstm_bf);

  gru_scan_kernel<<<B_ / 16, 512, 0, stream>>>(
      lstm_bf, gru_Wih, gru_Whh, gru_bih, gru_bhh, ln_gru_g, ln_gru_b, h_last);

  head_kernel<<<B_, 256, 0, stream>>>(
      h_last, fc1_W, fc1_b, ln_fc1_g, ln_fc1_b, fco_W, fco_b, out);
}
